// Attention_26877905338636
// MI455X (gfx1250) — matrix operand order repaired and checked by reading, not compiled, not run on hardware
//
#include <hip/hip_runtime.h>
#include <hip/hip_bf16.h>

typedef __bf16 bf16;
typedef __attribute__((ext_vector_type(16))) __bf16 v16bf;
typedef __attribute__((ext_vector_type(8)))  __bf16 v8bf;
typedef __attribute__((ext_vector_type(8)))  float  v8f;

#define DMODEL 4096
#define NH 64
#define HD 512
#define RD 64
#define QLR 1024
#define OLR 1024
#define OG 8
#define SEQ 1024
#define EPSF 1e-6f
#define KC 64                      // K elements staged per pipeline stage
#define TPAD 72                    // LDS tile row stride (64 + 8 pad, 144B, 16B-aligned)

static __device__ __forceinline__ v8f vzero() {
    v8f z = {0.f,0.f,0.f,0.f,0.f,0.f,0.f,0.f};
    return z;
}

static __device__ __forceinline__ v8f wmma_bf16(v16bf a, v16bf b, v8f c) {
    return __builtin_amdgcn_wmma_f32_16x16x32_bf16(
        /*neg_a=*/false, a, /*neg_b=*/false, b,
        /*c_mod=*/(short)0, c, /*reuse_a=*/false, /*reuse_b=*/false);
}

// One per-lane 16-byte async copy global -> LDS (ASYNCcnt-tracked).
static __device__ __forceinline__ void async_copy16(unsigned lds_off, const bf16* gptr) {
    asm volatile("global_load_async_to_lds_b128 %0, %1, off"
                 :: "v"(lds_off), "v"(gptr) : "memory");
}
static __device__ __forceinline__ void wait_async0() {
    asm volatile("s_wait_asynccnt 0x0" ::: "memory");
}

// A fragment (16x32 bf16, MxK): lane holds row m=lane&15; VGPRs 0-3 hold
// K = kbase..kbase+7, VGPRs 4-7 hold K = kbase+16..kbase+23, kbase=(lane>>4)*8.
static __device__ __forceinline__ v16bf load_frag_a(const bf16* base, int ld, int k, int lane) {
    int m  = lane & 15;
    int ko = (lane >> 4) * 8;
    const bf16* p = base + (size_t)m * ld + k + ko;
    union { v16bf v; v8bf h[2]; } u;
    u.h[0] = *(const v8bf*)(p);
    u.h[1] = *(const v8bf*)(p + 16);
    return u.v;
}

// B fragment (32x16 bf16, KxN) from an N-major buffer:
// lane holds column n=lane&15, K = kbase..kbase+7 and kbase+16..kbase+23, kbase=(lane>>4)*8.
static __device__ __forceinline__ v16bf load_frag_b(const bf16* base, int ld, int k, int lane) {
    int n  = lane & 15;
    int ko = (lane >> 4) * 8;
    const bf16* p = base + (size_t)n * ld + k + ko;
    union { v16bf v; v8bf h[2]; } u;
    u.h[0] = *(const v8bf*)(p);
    u.h[1] = *(const v8bf*)(p + 16);
    return u.v;
}

// Cooperatively stage a 128 x 32 bf16 tile (K cols k0..k0+31, half of KC) (global row stride ldg) into LDS.
// 512 chunks of 16B over 256 threads -> 2 async issues per thread.
static __device__ __forceinline__ void stage_tile(const bf16* gbase, int ldg, int k0,
                                                  bf16 (*lds)[TPAD], int tid) {
#pragma unroll
    for (int i = 0; i < 2; ++i) {
        int c  = tid + i * 256;            // 0..511
        int r  = c >> 2;                   // tile row 0..127
        int ch = c & 3;                    // 16B chunk within row (4 chunks = 32 elems, half of KC=64)
        const bf16* gp = gbase + (size_t)r * ldg + k0 + ch * 8;
        unsigned loff = (unsigned)(size_t)&lds[r][ch * 8];
        async_copy16(loff, gp);
    }
}

// ---------------------------------------------------------------------------
// C = A (MxK, lda) * B^T (B is NxK, ldb). Double-buffered async-LDS pipeline.
// Block = 256 threads = 8 waves as 4(M)x2(N); wave tile 32x64; block tile 128x128.
// ---------------------------------------------------------------------------
template <bool OUT_BF16>
__global__ void __launch_bounds__(256)
gemm_nt(const bf16* __restrict__ A, const bf16* __restrict__ B,
        void* __restrict__ Cv, int K, int lda, int ldb, int ldc)
{
    const int tid  = threadIdx.x;
    const int lane = tid & 31;
    const int wave = tid >> 5;
    const int wm = wave >> 1;                 // 0..3
    const int wn = wave & 1;                  // 0..1
    const int row0 = blockIdx.y * 128 + wm * 32;
    const int col0 = blockIdx.x * 128 + wn * 64;

    __shared__ __align__(16) bf16 sA[2][128][TPAD];
    __shared__ __align__(16) bf16 sB[2][128][TPAD];

    v8f acc[2][4];
#pragma unroll
    for (int i = 0; i < 2; ++i)
#pragma unroll
        for (int j = 0; j < 4; ++j) acc[i][j] = vzero();

    const bf16* Ablk = A + (size_t)(blockIdx.y * 128) * lda;
    const bf16* Bblk = B + (size_t)(blockIdx.x * 128) * ldb;

    stage_tile(Ablk, lda, 0, sA[0], tid);
    stage_tile(Bblk, ldb, 0, sB[0], tid);

    const int nk = K / KC;
    for (int kb = 0; kb < nk; ++kb) {
        wait_async0();                        // own stage-kb copies done
        __syncthreads();                      // all waves' copies visible;
                                              // all waves done with buf kb^1
        const int cur = kb & 1;
        if (kb + 1 < nk) {                    // overlap next-stage copy w/ compute
            stage_tile(Ablk, lda, (kb + 1) * KC, sA[cur ^ 1], tid);
            stage_tile(Bblk, ldb, (kb + 1) * KC, sB[cur ^ 1], tid);
        }
        const bf16* At = &sA[cur][wm * 32][0];
        const bf16* Bt = &sB[cur][wn * 64][0];
#pragma unroll
        for (int ks = 0; ks < KC; ks += 32) {
            v16bf a0 = load_frag_a(At,            TPAD, ks, lane);
            v16bf a1 = load_frag_a(At + 16 * TPAD, TPAD, ks, lane);
            v16bf b0 = load_frag_b(Bt,             TPAD, ks, lane);
            v16bf b1 = load_frag_b(Bt + 16 * TPAD, TPAD, ks, lane);
            v16bf b2 = load_frag_b(Bt + 32 * TPAD, TPAD, ks, lane);
            v16bf b3 = load_frag_b(Bt + 48 * TPAD, TPAD, ks, lane);
            acc[0][0] = wmma_bf16(a0, b0, acc[0][0]);
            acc[0][1] = wmma_bf16(a0, b1, acc[0][1]);
            acc[0][2] = wmma_bf16(a0, b2, acc[0][2]);
            acc[0][3] = wmma_bf16(a0, b3, acc[0][3]);
            acc[1][0] = wmma_bf16(a1, b0, acc[1][0]);
            acc[1][1] = wmma_bf16(a1, b1, acc[1][1]);
            acc[1][2] = wmma_bf16(a1, b2, acc[1][2]);
            acc[1][3] = wmma_bf16(a1, b3, acc[1][3]);
        }
    }

#pragma unroll
    for (int i = 0; i < 2; ++i) {
#pragma unroll
        for (int j = 0; j < 4; ++j) {
            int n  = col0 + j * 16 + (lane & 15);
            int mr = row0 + i * 16 + (lane >> 4) * 8;
#pragma unroll
            for (int r = 0; r < 8; ++r) {
                if (OUT_BF16)
                    ((bf16*)Cv)[(size_t)(mr + r) * ldc + n] = (bf16)acc[i][j][r];
                else
                    ((float*)Cv)[(size_t)(mr + r) * ldc + n] = acc[i][j][r];
            }
        }
    }
}

// ---------------------------------------------------------------------------
// rmsnorm over f32 rows -> bf16, with bf16 weight
// ---------------------------------------------------------------------------
__global__ void __launch_bounds__(256)
rmsnorm_kernel(const float* __restrict__ src, const bf16* __restrict__ w,
               bf16* __restrict__ dst, int N)
{
    int row = blockIdx.x;
    const float* p = src + (size_t)row * N;
    float s = 0.f;
    for (int i = threadIdx.x; i < N; i += 256) { float v = p[i]; s += v * v; }
    __shared__ float red[256];
    red[threadIdx.x] = s;
    __syncthreads();
    for (int off = 128; off > 0; off >>= 1) {
        if ((int)threadIdx.x < off) red[threadIdx.x] += red[threadIdx.x + off];
        __syncthreads();
    }
    float inv = rsqrtf(red[0] / (float)N + EPSF);
    for (int i = threadIdx.x; i < N; i += 256) {
        bf16 nb = (bf16)(p[i] * inv);
        dst[(size_t)row * N + i] = (bf16)((float)nb * (float)w[i]);
    }
}

// ---------------------------------------------------------------------------
// per-(token,head) q rmsnorm (no weight) + RoPE on last RD dims
// ---------------------------------------------------------------------------
__global__ void __launch_bounds__(256)
qnorm_rope_kernel(bf16* __restrict__ q, const float* __restrict__ cosv,
                  const float* __restrict__ sinv)
{
    int t = blockIdx.x, h = blockIdx.y;
    bf16* p = q + (size_t)t * (NH * HD) + (size_t)h * HD;
    float s = 0.f;
    for (int i = threadIdx.x; i < HD; i += 256) { float v = (float)p[i]; s += v * v; }
    __shared__ float red[256];
    red[threadIdx.x] = s;
    __syncthreads();
    for (int off = 128; off > 0; off >>= 1) {
        if ((int)threadIdx.x < off) red[threadIdx.x] += red[threadIdx.x + off];
        __syncthreads();
    }
    float inv = rsqrtf(red[0] / (float)HD + EPSF);
    for (int i = threadIdx.x; i < HD; i += 256)
        p[i] = (bf16)((float)p[i] * inv);
    __syncthreads();
    if (threadIdx.x < RD / 2) {
        int i = threadIdx.x;
        float c  = cosv[t * (RD / 2) + i];
        float sn = sinv[t * (RD / 2) + i];
        int b = HD - RD + 2 * i;
        float x1 = (float)p[b], x2 = (float)p[b + 1];
        p[b]     = (bf16)(x1 * c - x2 * sn);
        p[b + 1] = (bf16)(x1 * sn + x2 * c);
    }
}

// ---------------------------------------------------------------------------
// kv: rmsnorm(f32) * w -> bf16, then RoPE on last RD dims
// ---------------------------------------------------------------------------
__global__ void __launch_bounds__(256)
kv_post_kernel(const float* __restrict__ src, const bf16* __restrict__ w,
               const float* __restrict__ cosv, const float* __restrict__ sinv,
               bf16* __restrict__ kvout)
{
    int t = blockIdx.x;
    const float* p = src + (size_t)t * HD;
    float s = 0.f;
    for (int i = threadIdx.x; i < HD; i += 256) { float v = p[i]; s += v * v; }
    __shared__ float red[256];
    red[threadIdx.x] = s;
    __syncthreads();
    for (int off = 128; off > 0; off >>= 1) {
        if ((int)threadIdx.x < off) red[threadIdx.x] += red[threadIdx.x + off];
        __syncthreads();
    }
    float inv = rsqrtf(red[0] / (float)HD + EPSF);
    bf16* out = kvout + (size_t)t * HD;
    for (int i = threadIdx.x; i < HD; i += 256) {
        bf16 nb = (bf16)(p[i] * inv);
        out[i] = (bf16)((float)nb * (float)w[i]);
    }
    __syncthreads();
    if (threadIdx.x < RD / 2) {
        int i = threadIdx.x;
        float c  = cosv[t * (RD / 2) + i];
        float sn = sinv[t * (RD / 2) + i];
        int b = HD - RD + 2 * i;
        float x1 = (float)out[b], x2 = (float)out[b + 1];
        out[b]     = (bf16)(x1 * c - x2 * sn);
        out[b + 1] = (bf16)(x1 * sn + x2 * c);
    }
}

__global__ void __launch_bounds__(256)
kv_transpose_kernel(const bf16* __restrict__ kv, bf16* __restrict__ kvT)
{
    int idx = blockIdx.x * 256 + threadIdx.x;     // HD*SEQ
    if (idx >= HD * SEQ) return;
    int d = idx / SEQ, t = idx % SEQ;
    kvT[idx] = kv[(size_t)t * HD + d];
}

// ---------------------------------------------------------------------------
// attention: one workgroup per (16-query block, head); 8 waves
// ---------------------------------------------------------------------------
__global__ void __launch_bounds__(256)
attn_kernel(const bf16* __restrict__ q, const bf16* __restrict__ kv,
            const bf16* __restrict__ kvT, const float* __restrict__ sink,
            bf16* __restrict__ o)
{
    const int q0   = blockIdx.x * 16;
    const int head = blockIdx.y;
    const int tid  = threadIdx.x;
    const int lane = tid & 31;
    const int wave = tid >> 5;
    const float scale = 0.04419417382415922f;     // 512^-0.5

    __shared__ __align__(16) bf16 at[16][SEQ + 16];   // scores -> probs (bf16)
    __shared__ __align__(16) bf16 qs[16][HD + 8];     // staged q row-block

    // ---- phase 0: async-stage a 16x256 q tile (dims 0..255) into LDS (once) ----
    const bf16* qbase = q + (size_t)q0 * (NH * HD) + (size_t)head * HD;
#pragma unroll
    for (int i = 0; i < 2; ++i) {
        int c  = tid + i * 256;                   // 0..511
        int r  = c >> 5;                          // row 0..15
        int ch = c & 31;                          // 16B chunk (32 chunks = 256 of 512 elems)
        const bf16* gp = qbase + (size_t)r * (NH * HD) + ch * 8;
        unsigned loff = (unsigned)(size_t)&qs[r][ch * 8];
        async_copy16(loff, gp);
    }
    wait_async0();
    __syncthreads();

    // ---- phase 1: scores = q . kv^T, scaled + causal masked -> LDS ----
    const int m8 = (lane >> 4) * 8;
#pragma unroll 1
    for (int j = 0; j < 8; ++j) {
        int c0 = wave * 128 + j * 16;
        if (c0 > q0 + 15) {                       // tile fully above diagonal
#pragma unroll
            for (int r = 0; r < 8; ++r)
                at[m8 + r][c0 + (lane & 15)] = (bf16)(-__builtin_inff());
            continue;
        }
        v8f acc = vzero();
        for (int kd = 0; kd < HD; kd += 32) {
            v16bf a = load_frag_a(&qs[0][0], HD + 8, kd, lane);
            v16bf b = load_frag_b(kv + (size_t)c0 * HD, HD, kd, lane);
            acc = wmma_bf16(a, b, acc);
        }
        int n = c0 + (lane & 15);
#pragma unroll
        for (int r = 0; r < 8; ++r) {
            int m = m8 + r;
            float sv = (n <= q0 + m) ? acc[r] * scale : -__builtin_inff();
            at[m][n] = (bf16)sv;
        }
    }
    __syncthreads();

    // ---- phase 2: sink-augmented softmax (wave handles 2 rows) ----
    float snk = sink[head];
#pragma unroll 1
    for (int m = wave * 2; m < wave * 2 + 2; ++m) {
        float lmx = -__builtin_inff();
        for (int c = lane; c < SEQ; c += 32) lmx = fmaxf(lmx, (float)at[m][c]);
        for (int off = 16; off > 0; off >>= 1) lmx = fmaxf(lmx, __shfl_xor(lmx, off, 32));
        float mx = fmaxf(lmx, snk);
        float lsum = 0.f;
        for (int c = lane; c < SEQ; c += 32) {
            float e = __expf((float)at[m][c] - mx);   // exp(-inf)=0 handles mask
            at[m][c] = (bf16)e;
            lsum += e;
        }
        for (int off = 16; off > 0; off >>= 1) lsum += __shfl_xor(lsum, off, 32);
        float inv = 1.f / (lsum + __expf(snk - mx));  // sink in denominator
        for (int c = lane; c < SEQ; c += 32)
            at[m][c] = (bf16)((float)at[m][c] * inv);
    }
    __syncthreads();

    // ---- phase 3: o = probs . kv  (bounded by causal extent) ----
    int klen = (q0 + 16 + 31) & ~31;
    if (klen > SEQ) klen = SEQ;
    const bf16* atp = &at[0][0];
#pragma unroll 1
    for (int j = 0; j < 4; ++j) {
        int n0 = wave * 64 + j * 16;              // head-dim column base
        v8f acc = vzero();
        for (int kt = 0; kt < klen; kt += 32) {
            v16bf a = load_frag_a(atp, SEQ + 16, kt, lane);     // from LDS
            v16bf b = load_frag_b(kvT + (size_t)n0 * SEQ, SEQ, kt, lane);
            acc = wmma_bf16(a, b, acc);
        }
        int nh = n0 + (lane & 15);
#pragma unroll
        for (int r = 0; r < 8; ++r) {
            int m = m8 + r;
            o[(size_t)(q0 + m) * (NH * HD) + (size_t)head * HD + nh] = (bf16)acc[r];
        }
    }
}

// ---------------------------------------------------------------------------
// inverse RoPE on last RD dims of o, per (token, head)
// ---------------------------------------------------------------------------
__global__ void __launch_bounds__(32)
o_invrope_kernel(bf16* __restrict__ o, const float* __restrict__ cosv,
                 const float* __restrict__ sinv)
{
    int t = blockIdx.x, h = blockIdx.y;
    int i = threadIdx.x;                          // 0..31
    bf16* p = o + (size_t)t * (NH * HD) + (size_t)h * HD + (HD - RD);
    float c  = cosv[t * (RD / 2) + i];
    float sn = -sinv[t * (RD / 2) + i];           // inverse rotation
    float x1 = (float)p[2 * i], x2 = (float)p[2 * i + 1];
    p[2 * i]     = (bf16)(x1 * c - x2 * sn);
    p[2 * i + 1] = (bf16)(x1 * sn + x2 * c);
}

// ---------------------------------------------------------------------------
extern "C" void kernel_launch(void* const* d_in, const int* in_sizes, int n_in,
                              void* d_out, int out_size, void* d_ws, size_t ws_size,
                              hipStream_t stream) {
    (void)in_sizes; (void)n_in; (void)out_size; (void)ws_size;
    const bf16*  x         = (const bf16*)d_in[0];
    const bf16*  wq_a      = (const bf16*)d_in[1];
    const bf16*  q_norm_w  = (const bf16*)d_in[2];
    const bf16*  wq_b      = (const bf16*)d_in[3];
    const bf16*  wkv       = (const bf16*)d_in[4];
    const bf16*  kv_norm_w = (const bf16*)d_in[5];
    const bf16*  wo_a      = (const bf16*)d_in[6];
    const bf16*  wo_b      = (const bf16*)d_in[7];
    const float* sink      = (const float*)d_in[8];
    const float* cosv      = (const float*)d_in[9];
    const float* sinv      = (const float*)d_in[10];

    char* ws = (char*)d_ws;
    float* tmp1  = (float*)(ws);                          //  4 MB (1024x1024 f32)
    float* kvtmp = (float*)(ws + (size_t)( 4 << 20));     //  2 MB (1024x512 f32)
    bf16*  qr    = (bf16*) (ws + (size_t)( 6 << 20));     //  2 MB
    bf16*  kvb   = (bf16*) (ws + (size_t)( 8 << 20));     //  1 MB
    bf16*  kvT   = (bf16*) (ws + (size_t)( 9 << 20));     //  1 MB
    bf16*  qbuf  = (bf16*) (ws + (size_t)(10 << 20));     // 64 MB (1024x32768)
    bf16*  obuf  = (bf16*) (ws + (size_t)(74 << 20));     // 64 MB
    bf16*  ogb   = (bf16*) (ws + (size_t)(138 << 20));    // 16 MB

    dim3 blk(256);

    // 1) qr = rms(x @ wq_a^T) * q_norm_w
    gemm_nt<false><<<dim3(QLR / 128, SEQ / 128), blk, 0, stream>>>(
        x, wq_a, tmp1, DMODEL, DMODEL, DMODEL, QLR);
    rmsnorm_kernel<<<SEQ, blk, 0, stream>>>(tmp1, q_norm_w, qr, QLR);

    // 2) q = qr @ wq_b^T ; per-head rmsnorm + rope
    gemm_nt<true><<<dim3((NH * HD) / 128, SEQ / 128), blk, 0, stream>>>(
        qr, wq_b, qbuf, QLR, QLR, QLR, NH * HD);
    qnorm_rope_kernel<<<dim3(SEQ, NH), blk, 0, stream>>>(qbuf, cosv, sinv);

    // 3) kv = rms(x @ wkv^T) * kv_norm_w + rope ; transpose for PV GEMM
    gemm_nt<false><<<dim3(HD / 128, SEQ / 128), blk, 0, stream>>>(
        x, wkv, kvtmp, DMODEL, DMODEL, DMODEL, HD);
    kv_post_kernel<<<SEQ, blk, 0, stream>>>(kvtmp, kv_norm_w, cosv, sinv, kvb);
    kv_transpose_kernel<<<(HD * SEQ + 255) / 256, blk, 0, stream>>>(kvb, kvT);

    // 4) attention (causal + sink softmax)
    attn_kernel<<<dim3(SEQ / 16, NH), blk, 0, stream>>>(qbuf, kvb, kvT, sink, obuf);

    // 5) inverse rope on o
    o_invrope_kernel<<<dim3(SEQ, NH), dim3(32), 0, stream>>>(obuf, cosv, sinv);

    // 6) grouped low-rank out-projection: og[:, g*1024:(g+1)*1024]
    for (int g = 0; g < OG; ++g) {
        gemm_nt<true><<<dim3(OLR / 128, SEQ / 128), blk, 0, stream>>>(
            obuf + (size_t)g * (NH * HD / OG),
            wo_a + (size_t)g * OLR * (NH * HD / OG),
            ogb  + (size_t)g * OLR,
            NH * HD / OG, NH * HD, NH * HD / OG, OG * OLR);
    }

    // 7) out = og @ wo_b^T
    gemm_nt<true><<<dim3(DMODEL / 128, SEQ / 128), blk, 0, stream>>>(
        ogb, wo_b, (bf16*)d_out, OG * OLR, OG * OLR, OG * OLR, DMODEL);
}
